// GAT_GRU_47132971106433
// MI455X (gfx1250) — matrix operand order repaired and checked by reading, not compiled, not run on hardware
//
#include <hip/hip_runtime.h>
#include <cmath>

typedef __attribute__((ext_vector_type(16))) _Float16 v16h;
typedef __attribute__((ext_vector_type(8)))  float    v8f;

// Problem constants (from reference)
constexpr int Tdim  = 8;
constexpr int Nn    = 10000;
constexpr int Ee    = 160000;
constexpr int EpE   = Ee + Nn;          // edges + self loops = 170000
constexpr int HEADS = 4;
constexpr int ROWS  = Tdim * Nn;        // 80000, multiple of 16

// ---------------------------------------------------------------------------
// Utility kernels
// ---------------------------------------------------------------------------
__global__ void k_fill(float* __restrict__ p, float v, int n) {
  int i = blockIdx.x * blockDim.x + threadIdx.x;
  if (i < n) p[i] = v;
}

__device__ inline void atomicMaxF(float* addr, float val) {
  unsigned int* ua = reinterpret_cast<unsigned int*>(addr);
  unsigned int old = *ua;
  while (true) {
    float f = __uint_as_float(old);
    if (f >= val) break;
    unsigned int assumed = old;
    old = atomicCAS(ua, assumed, __float_as_uint(val));
    if (old == assumed) break;
  }
}

// ---------------------------------------------------------------------------
// WMMA GEMM: C[M,NCOL] = A[M,K] * B[K,NCOL], fp32 in/out, f16 multiply,
// f32 accumulate. One wave32 computes one 16x16 output tile.
// K, NCOL are compile-time: all fragment bounds checks fold away.
// B is staged into LDS (converted to f16, zero-padded to Kpad rows).
// ---------------------------------------------------------------------------
template <int K, int NCOL>
__global__ void k_gemm_wmma(const float* __restrict__ A, const float* __restrict__ B,
                            float* __restrict__ C, int M) {
  constexpr int Kpad = (K + 31) & ~31;          // K rounded up to multiple of 32
  __shared__ _Float16 sB[Kpad * NCOL];

  // Cooperative stage + f32->f16 convert of the weight matrix; pad rows = 0.
  for (int i = threadIdx.x; i < Kpad * NCOL; i += blockDim.x)
    sB[i] = (i < K * NCOL) ? (_Float16)B[i] : (_Float16)0.0f;
  __syncthreads();

  const int lane        = threadIdx.x & 31;
  const int waveInBlk   = threadIdx.x >> 5;
  const int wavesPerBlk = blockDim.x >> 5;
  const int wave        = blockIdx.x * wavesPerBlk + waveInBlk;
  constexpr int colTiles = NCOL / 16;
  const int rowTile = wave / colTiles;
  const int colTile = wave - rowTile * colTiles;
  if (rowTile * 16 >= M) return;                // wave-uniform

  const int hi  = lane >> 4;                    // 0 = lanes 0-15, 1 = lanes 16-31
  const int l16 = lane & 15;
  const int kbA = hi * 8;                       // A K-base per ISA 16-bit A layout
  const int kbB = hi * 8;                       // B K-base per ISA 16-bit B layout
  const int col = colTile * 16 + l16;           // output column
  const float* __restrict__ arowp = A + (size_t)(rowTile * 16 + l16) * K;

  v8f acc = {};
#pragma unroll
  for (int kk = 0; kk < K; kk += 32) {
    union { v16h v; _Float16 e[16]; } a, b;
    // A fragment: elems 0..7 -> K = kk+kbA+0..7 (always < K);
    //             elems 8..15 -> K = kk+16+kbA+0..7 (only valid when K%32==0)
#pragma unroll
    for (int i = 0; i < 8; ++i) a.e[i] = (_Float16)arowp[kk + kbA + i];
    if constexpr ((K % 32) == 0) {
#pragma unroll
      for (int i = 0; i < 8; ++i) a.e[8 + i] = (_Float16)arowp[kk + 16 + kbA + i];
    } else {
#pragma unroll
      for (int i = 0; i < 8; ++i) a.e[8 + i] = (_Float16)0.0f;
    }
    // B fragment from LDS (zero-padded, so no bounds checks)
#pragma unroll
    for (int i = 0; i < 16; ++i) b.e[i] = sB[(kk + kbB + (i & 7) + (i >> 3) * 16) * NCOL + col];

    acc = __builtin_amdgcn_wmma_f32_16x16x32_f16(false, a.v, false, b.v,
                                                 (short)0, acc, false, false);
  }

  union { v8f v; float e[8]; } co;
  co.v = acc;
  const int mrow = rowTile * 16 + hi * 8;       // C/D: VGPR r -> M = r (+8 upper)
#pragma unroll
  for (int r = 0; r < 8; ++r) C[(mrow + r) * NCOL + col] = co.e[r];
}

// ---------------------------------------------------------------------------
// asrc/adst = sum_c h[row, head, c] * a_{src,dst}[head, c]
// ---------------------------------------------------------------------------
__global__ void k_attn_coef(const float* __restrict__ h,
                            const float* __restrict__ a_src, const float* __restrict__ a_dst,
                            float* __restrict__ asrc, float* __restrict__ adst, int C) {
  int idx = blockIdx.x * blockDim.x + threadIdx.x;   // ROWS*HEADS
  if (idx >= ROWS * HEADS) return;
  int row = idx >> 2;
  int hh  = idx & 3;
  const float* hp = h + (row * HEADS + hh) * C;
  float s1 = 0.f, s2 = 0.f;
  for (int c = 0; c < C; ++c) {
    float v = hp[c];
    s1 += v * a_src[hh * C + c];
    s2 += v * a_dst[hh * C + c];
  }
  asrc[idx] = s1;
  adst[idx] = s2;
}

__device__ inline void edge_nodes(const long long* __restrict__ ei, int e, int& s, int& d) {
  if (e < Ee) { s = (int)ei[e]; d = (int)ei[Ee + e]; }
  else        { s = d = e - Ee; }        // appended self loops
}

// score = leaky_relu(asrc[t,src,h] + adst[t,dst,h]); running segment-max into m
__global__ void k_edge_score(const long long* __restrict__ ei,
                             const float* __restrict__ asrc, const float* __restrict__ adst,
                             float* __restrict__ esc, float* __restrict__ m) {
  int idx = blockIdx.x * blockDim.x + threadIdx.x;
  if (idx >= EpE * Tdim * HEADS) return;
  int e  = idx >> 5;            // Tdim*HEADS == 32
  int th = idx & 31;
  int t  = th >> 2;
  int hh = th & 3;
  int s, d;
  edge_nodes(ei, e, s, d);
  float sc = asrc[(t * Nn + s) * HEADS + hh] + adst[(t * Nn + d) * HEADS + hh];
  sc = sc > 0.f ? sc : 0.2f * sc;
  esc[idx] = sc;
  atomicMaxF(&m[(d * Tdim + t) * HEADS + hh], sc);
}

// e = exp(score - m[dst]); denom[dst] += e
__global__ void k_edge_exp(const long long* __restrict__ ei,
                           float* __restrict__ esc, const float* __restrict__ m,
                           float* __restrict__ denom) {
  int idx = blockIdx.x * blockDim.x + threadIdx.x;
  if (idx >= EpE * Tdim * HEADS) return;
  int e  = idx >> 5;
  int th = idx & 31;
  int t  = th >> 2;
  int hh = th & 3;
  int s, d;
  edge_nodes(ei, e, s, d);
  float v = expf(esc[idx] - m[(d * Tdim + t) * HEADS + hh]);
  esc[idx] = v;
  atomicAdd(&denom[(d * Tdim + t) * HEADS + hh], v);
}

// agg[dst] += alpha * h[src]
__global__ void k_edge_agg(const long long* __restrict__ ei,
                           const float* __restrict__ esc, const float* __restrict__ denom,
                           const float* __restrict__ hfeat, float* __restrict__ agg, int C) {
  int idx = blockIdx.x * blockDim.x + threadIdx.x;
  if (idx >= EpE * Tdim * HEADS) return;
  int e  = idx >> 5;
  int th = idx & 31;
  int t  = th >> 2;
  int hh = th & 3;
  int s, d;
  edge_nodes(ei, e, s, d);
  float alpha = esc[idx] / (denom[(d * Tdim + t) * HEADS + hh] + 1e-16f);
  const float* hs = hfeat + ((t * Nn + s) * HEADS + hh) * C;
  float* ag = agg + ((d * Tdim + t) * HEADS + hh) * C;
  for (int c = 0; c < C; ++c) atomicAdd(&ag[c], alpha * hs[c]);
}

// out[(t*N+n), f] = relu(agg[n,t,head(f),chan(f)] + bias[f])   (F = HEADS*C)
__global__ void k_bias_relu(const float* __restrict__ agg, const float* __restrict__ bias,
                            float* __restrict__ outp, int C) {
  int F = HEADS * C;
  int idx = blockIdx.x * blockDim.x + threadIdx.x;
  if (idx >= ROWS * F) return;
  int row = idx / F;
  int f   = idx - row * F;
  int t   = row / Nn;
  int n   = row - t * Nn;
  int hh  = f / C;
  int cc  = f - hh * C;
  float v = agg[((n * Tdim + t) * HEADS + hh) * C + cc] + bias[f];
  outp[idx] = v > 0.f ? v : 0.f;
}

// ---------------------------------------------------------------------------
// GRU scan: seq = N (10000), batch = 32, input = 8 (=Tdim), hidden = 32.
// Single persistent 1024-thread workgroup (32 wave32s); state + weights in LDS.
// out[HID, N, 32]: out[j*N*32 + n*32 + b]
// ---------------------------------------------------------------------------
__global__ void __launch_bounds__(1024) k_gru(const float* __restrict__ hfin,
                                              const float* __restrict__ w_ih,
                                              const float* __restrict__ w_hh,
                                              const float* __restrict__ b_ih,
                                              const float* __restrict__ b_hh,
                                              float* __restrict__ out) {
  __shared__ float hprev[32 * 32];
  __shared__ float xs[32 * 8];
  __shared__ float s_wih[96 * 8];
  __shared__ float s_whh[96 * 32];
  __shared__ float s_bih[96];
  __shared__ float s_bhh[96];
  const int tid = threadIdx.x;
  const int b = tid >> 5;      // batch index 0..31
  const int j = tid & 31;      // hidden index 0..31
  for (int i = tid; i < 96 * 8;  i += 1024) s_wih[i] = w_ih[i];
  for (int i = tid; i < 96 * 32; i += 1024) s_whh[i] = w_hh[i];
  if (tid < 96) { s_bih[tid] = b_ih[tid]; s_bhh[tid] = b_hh[tid]; }
  hprev[tid] = 0.f;
  __syncthreads();

  for (int n = 0; n < Nn; ++n) {
    if (j < 8) xs[b * 8 + j] = hfin[(j * Nn + n) * 32 + b];   // g[n,b,t]=hfin[t,n,b]
    __syncthreads();
    float ir = s_bih[j], iz = s_bih[32 + j], inn = s_bih[64 + j];
#pragma unroll
    for (int k = 0; k < 8; ++k) {
      float xv = xs[b * 8 + k];
      ir  += xv * s_wih[j * 8 + k];
      iz  += xv * s_wih[(32 + j) * 8 + k];
      inn += xv * s_wih[(64 + j) * 8 + k];
    }
    float hr = s_bhh[j], hz = s_bhh[32 + j], hn = s_bhh[64 + j];
#pragma unroll 8
    for (int k = 0; k < 32; ++k) {
      float hv = hprev[b * 32 + k];
      hr += hv * s_whh[j * 32 + k];
      hz += hv * s_whh[(32 + j) * 32 + k];
      hn += hv * s_whh[(64 + j) * 32 + k];
    }
    float r  = 1.f / (1.f + expf(-(ir + hr)));
    float z  = 1.f / (1.f + expf(-(iz + hz)));
    float nv = tanhf(inn + r * hn);
    float hp = hprev[b * 32 + j];
    float hnew = (1.f - z) * nv + z * hp;
    __syncthreads();
    hprev[b * 32 + j] = hnew;
    out[(j * Nn + n) * 32 + b] = hnew;
    __syncthreads();
  }
}

// ---------------------------------------------------------------------------
// Host launcher
// ---------------------------------------------------------------------------
extern "C" void kernel_launch(void* const* d_in, const int* in_sizes, int n_in,
                              void* d_out, int out_size, void* d_ws, size_t ws_size,
                              hipStream_t stream) {
  const float*     x      = (const float*)d_in[0];
  const long long* ei     = (const long long*)d_in[1];
  const float*     W1     = (const float*)d_in[2];
  const float*     a_src1 = (const float*)d_in[3];
  const float*     a_dst1 = (const float*)d_in[4];
  const float*     b1     = (const float*)d_in[5];
  const float*     W2     = (const float*)d_in[6];
  const float*     a_src2 = (const float*)d_in[7];
  const float*     a_dst2 = (const float*)d_in[8];
  const float*     b2     = (const float*)d_in[9];
  const float*     w_ih   = (const float*)d_in[10];
  const float*     w_hh   = (const float*)d_in[11];
  const float*     b_ih   = (const float*)d_in[12];
  const float*     b_hh   = (const float*)d_in[13];
  float* out = (float*)d_out;

  // workspace layout (floats)
  float* ws   = (float*)d_ws;
  float* h1   = ws;                       // 80000*128 = 10,240,000 (reused as conv2 input)
  float* asrc = h1   + 10240000;          //   320,000
  float* adst = asrc + 320000;            //   320,000
  float* mm   = adst + 320000;            //   320,000
  float* den  = mm   + 320000;            //   320,000
  float* esc  = den  + 320000;            // 5,440,000
  float* agg1 = esc  + 5440000;           // 10,240,000
  float* h2   = agg1 + 10240000;          // 2,560,000
  float* agg2 = h2   + 2560000;           // 2,560,000
  float* hfin = agg2 + 2560000;           // 2,560,000

  const int thr = 256;
  auto cdiv = [](int a, int b) { return (a + b - 1) / b; };
  const int NTH   = Nn * Tdim * HEADS;          // 320000
  const int EDGEW = EpE * Tdim * HEADS;         // 5,440,000

  // ---- conv1 ----
  k_fill<<<cdiv(NTH, thr), thr, 0, stream>>>(mm, -1e30f, NTH);
  k_fill<<<cdiv(NTH, thr), thr, 0, stream>>>(den, 0.f, NTH);
  k_fill<<<cdiv(NTH * 32, thr), thr, 0, stream>>>(agg1, 0.f, NTH * 32);
  // 5000 row tiles x 8 col tiles = 40000 waves, 8 waves/block
  k_gemm_wmma<16, 128><<<5000, 256, 0, stream>>>(x, W1, h1, ROWS);
  k_attn_coef<<<cdiv(ROWS * HEADS, thr), thr, 0, stream>>>(h1, a_src1, a_dst1, asrc, adst, 32);
  k_edge_score<<<cdiv(EDGEW, thr), thr, 0, stream>>>(ei, asrc, adst, esc, mm);
  k_edge_exp<<<cdiv(EDGEW, thr), thr, 0, stream>>>(ei, esc, mm, den);
  k_edge_agg<<<cdiv(EDGEW, thr), thr, 0, stream>>>(ei, esc, den, h1, agg1, 32);
  k_bias_relu<<<cdiv(ROWS * 128, thr), thr, 0, stream>>>(agg1, b1, h1, 32); // h1 now conv2 input

  // ---- conv2 ----
  k_fill<<<cdiv(NTH, thr), thr, 0, stream>>>(mm, -1e30f, NTH);
  k_fill<<<cdiv(NTH, thr), thr, 0, stream>>>(den, 0.f, NTH);
  k_fill<<<cdiv(NTH * 8, thr), thr, 0, stream>>>(agg2, 0.f, NTH * 8);
  // 5000 row tiles x 2 col tiles = 10000 waves
  k_gemm_wmma<128, 32><<<1250, 256, 0, stream>>>(h1, W2, h2, ROWS);
  k_attn_coef<<<cdiv(ROWS * HEADS, thr), thr, 0, stream>>>(h2, a_src2, a_dst2, asrc, adst, 8);
  k_edge_score<<<cdiv(EDGEW, thr), thr, 0, stream>>>(ei, asrc, adst, esc, mm);
  k_edge_exp<<<cdiv(EDGEW, thr), thr, 0, stream>>>(ei, esc, mm, den);
  k_edge_agg<<<cdiv(EDGEW, thr), thr, 0, stream>>>(ei, esc, den, h2, agg2, 8);
  k_bias_relu<<<cdiv(ROWS * 32, thr), thr, 0, stream>>>(agg2, b2, hfin, 8);

  // ---- GRU scan (persistent single workgroup) ----
  k_gru<<<1, 1024, 0, stream>>>(hfin, w_ih, w_hh, b_ih, b_hh, out);
}
